// GVP_Model_19138374271328
// MI455X (gfx1250) — matrix operand order repaired and checked by reading, not compiled, not run on hardware
//
#include <hip/hip_runtime.h>

// ---------------------------------------------------------------------------
// Types for CDNA5 WMMA / TDM
// ---------------------------------------------------------------------------
typedef __attribute__((ext_vector_type(16))) __bf16 v16bf;
typedef __attribute__((ext_vector_type(8)))  __bf16 v8bf;
typedef __attribute__((ext_vector_type(8)))  float  v8f;
typedef __attribute__((ext_vector_type(4))) unsigned int tdm_u4;
typedef __attribute__((ext_vector_type(8))) int          tdm_i8;
typedef __attribute__((ext_vector_type(4))) int          tdm_i4;

#define EPSF 1e-8f

#if defined(__has_builtin)
#  if __has_builtin(__builtin_amdgcn_tensor_load_to_lds) && \
      __has_builtin(__builtin_amdgcn_s_wait_tensorcnt)
#    define HAVE_TDM 1
#  endif
#endif
#ifndef HAVE_TDM
#  define HAVE_TDM 0
#endif

static __device__ __forceinline__ v16bf mk16(v8bf lo, v8bf hi) {
  v16bf r;
  #pragma unroll
  for (int i = 0; i < 8; ++i) { r[i] = lo[i]; r[8 + i] = hi[i]; }
  return r;
}

// ---------------------------------------------------------------------------
// B-operand prep: f32 (K x N) -> bf16 (Np x Kp) transposed + zero padded,
// so a block's 64-column chunk is contiguous and TDM/LDS friendly.
// ---------------------------------------------------------------------------
__global__ void prep_B_bf16(const float* __restrict__ W, __bf16* __restrict__ Bt,
                            int K, int N, int Kp, long total) {
  long idx = blockIdx.x * (long)blockDim.x + threadIdx.x;
  if (idx >= total) return;
  int n = (int)(idx / Kp), k = (int)(idx % Kp);
  Bt[idx] = (k < K && n < N) ? (__bf16)W[(long)k * N + n] : (__bf16)0.f;
}

// ---------------------------------------------------------------------------
// WMMA GEMM:  C[M x N] = act( A[M x lda](bf16) @ B[K x N] + bias )
//  * B given pre-transposed/padded as Bt (Np x Kp);  block's 64-col chunk is
//    staged into LDS ONCE (TDM tensor_load_to_lds when available), laid out
//    [col][K] so a lane's B fragment = 2 contiguous ds_load_b128.
//  * A fragments are loaded straight from global: per the ISA layout a lane
//    needs two contiguous 16B runs of its own row -> 2 global_load_b128.
//  * 8 waves/block, each wave owns 16 rows x 64 cols = 4 accumulators; the
//    K-loop has NO barriers: 2 vmem loads feed 4 v_wmma per iteration.
// ---------------------------------------------------------------------------
__global__ __launch_bounds__(256)
void gemm_bf16_wmma(const __bf16* __restrict__ A, int lda,
                    const __bf16* __restrict__ Bt,
                    const float* __restrict__ bias,
                    float* __restrict__ C, int ldc,
                    long M, int K, int N, int Kp, int act) {
  extern __shared__ __align__(16) char smem[];
  __bf16* lB = (__bf16*)smem;                       // [64][Kp]

  const int tid  = threadIdx.x;
  const int wave = tid >> 5;
  const int lane = tid & 31;
  const int half = lane >> 4;
  const int r16  = lane & 15;

  const long mBase = (long)blockIdx.x * 128;
  const int  nc    = blockIdx.y * 64;
  const long arow  = mBase + wave * 16 + r16;       // row this lane feeds into A frags
  const bool rowOk = arow < M;
  const __bf16* Ap = A + arow * (long)lda;

  // ---- stage the 64 x Kp B panel into LDS (once per block)
  const __bf16* bsrc = Bt + (long)nc * Kp;
#if HAVE_TDM
  if (wave == 0) {
    unsigned ldsAddr = (unsigned)(size_t)(void*)lB; // low 32 bits = LDS offset
    unsigned long long ga = (unsigned long long)(size_t)(const void*)bsrc;
    tdm_u4 g0 = { 1u,                               // count=1, user mode
                  ldsAddr,
                  (unsigned)ga,
                  ((unsigned)(ga >> 32) & 0x1FFFFFFu) | (2u << 30) }; // type=2
    tdm_i8 g1 = { (int)(1u << 16),                  // data_size = 2 bytes
                  (int)((unsigned)(Kp & 0xFFFF) << 16),   // tensor_dim0 lo
                  (int)(64u << 16),                 // tensor_dim0 hi=0 | tensor_dim1=64
                  (int)((unsigned)(Kp & 0xFFFF) << 16),   // dim1 hi=0 | tile_dim0=Kp
                  (int)64,                          // tile_dim1=64, tile_dim2=0
                  (int)Kp,                          // tensor_dim0_stride lo
                  0, 0 };
    tdm_i4 gz = { 0, 0, 0, 0 };
#if defined(__clang_major__) && (__clang_major__ >= 23)
    tdm_i8 gz8 = { 0, 0, 0, 0, 0, 0, 0, 0 };
    __builtin_amdgcn_tensor_load_to_lds(g0, g1, gz, gz, gz8, 0);
#else
    __builtin_amdgcn_tensor_load_to_lds(g0, g1, gz, gz, 0);
#endif
    __builtin_amdgcn_s_wait_tensorcnt(0);
  }
#else
  {
    const uint4* s4 = (const uint4*)bsrc;
    uint4*       d4 = (uint4*)lB;
    const int n4 = (64 * Kp) >> 3;
    for (int i = tid; i < n4; i += 256) d4[i] = s4[i];
  }
#endif
  __syncthreads();

  v8f acc[4] = {{}, {}, {}, {}};

  for (int k0 = 0; k0 < K; k0 += 32) {
    v8bf a0 = {}, a1 = {};
    if (rowOk) {
      a0 = *(const v8bf*)(Ap + k0 + 8 * half);
      a1 = *(const v8bf*)(Ap + k0 + 16 + 8 * half);
      __builtin_prefetch(Ap + k0 + 64, 0, 1);
    }
    v16bf af = mk16(a0, a1);
    #pragma unroll
    for (int t = 0; t < 4; ++t) {
      const __bf16* p = lB + (long)(t * 16 + r16) * Kp + (k0 + 8 * half);
      v16bf bf = mk16(*(const v8bf*)p, *(const v8bf*)(p + 16));
      acc[t] = __builtin_amdgcn_wmma_f32_16x16x32_bf16(false, af, false, bf,
                                                       (short)0, acc[t], false, false);
    }
  }

  // ---- epilogue: C row = mBase + wave*16 + r + 8*half, col = nc + t*16 + r16
  #pragma unroll
  for (int t = 0; t < 4; ++t) {
    int col = nc + t * 16 + r16;
    if (col < N) {
      float bv = (bias != nullptr) ? bias[col] : 0.f;
      #pragma unroll
      for (int r = 0; r < 8; ++r) {
        long orow = mBase + wave * 16 + r + 8 * half;
        if (orow < M) {
          float v = acc[t][r] + bv;
          if (act) v = fmaxf(v, 0.f);
          C[orow * (long)ldc + col] = v;
        }
      }
    }
  }
}

// ---------------------------------------------------------------------------
// Vector-channel matmul: Y[m,d,:] = X[m,d,:] @ W(vi x h), plus
// vn[m,j] = sqrt(clip(sum_d Y^2, EPS)).  Strided X so raw (N,vi,3) inputs
// can be consumed transposed.
// ---------------------------------------------------------------------------
__global__ void vchan_mm(const float* __restrict__ X, long sm, long sd, long sk,
                         const float* __restrict__ W,
                         float* __restrict__ Y, float* __restrict__ vn,
                         long M, int vi, int h) {
  long idx = blockIdx.x * (long)blockDim.x + threadIdx.x;
  if (idx >= M * h) return;
  long m = idx / h;
  int  j = (int)(idx % h);
  const float* xm = X + m * sm;
  float y0 = 0.f, y1 = 0.f, y2 = 0.f;
  for (int k = 0; k < vi; ++k) {
    float w = W[k * h + j];
    y0 += xm[0 * sd + k * sk] * w;
    y1 += xm[1 * sd + k * sk] * w;
    y2 += xm[2 * sd + k * sk] * w;
  }
  float* ym = Y + (m * 3) * (long)h + j;
  ym[0] = y0; ym[h] = y1; ym[2 * h] = y2;
  if (vn) vn[m * h + j] = sqrtf(fmaxf(y0 * y0 + y1 * y1 + y2 * y2, EPSF));
}

// Y = X(M,3,h) @ W(h x vo), optional sigmoid(norm) gate on each vector.
__global__ void vchan_out(const float* __restrict__ X, const float* __restrict__ W,
                          float* __restrict__ Y, long M, int h, int vo, int gate) {
  long idx = blockIdx.x * (long)blockDim.x + threadIdx.x;
  if (idx >= M * vo) return;
  long m = idx / vo;
  int  j = (int)(idx % vo);
  const float* xm = X + m * 3 * (long)h;
  float y0 = 0.f, y1 = 0.f, y2 = 0.f;
  for (int k = 0; k < h; ++k) {
    float w = W[k * vo + j];
    y0 += xm[k] * w; y1 += xm[h + k] * w; y2 += xm[2 * h + k] * w;
  }
  if (gate) {
    float nr = sqrtf(fmaxf(y0 * y0 + y1 * y1 + y2 * y2, EPSF));
    float g = 1.f / (1.f + expf(-nr));
    y0 *= g; y1 *= g; y2 *= g;
  }
  float* ym = Y + m * 3 * (long)vo + j;
  ym[0] = y0; ym[vo] = y1; ym[2 * vo] = y2;
}

// ---------------------------------------------------------------------------
// Edge gather/concat kernels (emit zero-padded bf16 GEMM panels)
// ---------------------------------------------------------------------------
__global__ void build_msg_mv(const int* __restrict__ src, const int* __restrict__ dst,
                             const float* __restrict__ hv, const float* __restrict__ encv,
                             const float* __restrict__ ev,
                             float* __restrict__ mv, long E) {
  long idx = blockIdx.x * (long)blockDim.x + threadIdx.x;
  if (idx >= E * 99) return;
  long e = idx / 99;
  int  r = (int)(idx % 99);
  int  d = r / 33, c = r % 33;
  int  s = src[e], t = dst[e];
  const float* h = (encv != nullptr && !(s < t)) ? encv : hv;
  float v;
  if (c < 16)       v = h[((long)s * 3 + d) * 16 + c];
  else if (c == 16) v = ev[e * 3 + d];
  else              v = h[((long)t * 3 + d) * 16 + (c - 17)];
  mv[idx] = v;
}

__global__ void build_msg_A(const int* __restrict__ src, const int* __restrict__ dst,
                            const float* __restrict__ hs, const float* __restrict__ encs,
                            const float* __restrict__ es, int se,
                            const float* __restrict__ vn1,
                            __bf16* __restrict__ A, int Kp, long E) {
  long idx = blockIdx.x * (long)blockDim.x + threadIdx.x;
  if (idx >= E * Kp) return;
  long e = idx / Kp;
  int  k = (int)(idx % Kp);
  int  s = src[e], t = dst[e];
  const float* h = (encs != nullptr && !(s < t)) ? encs : hs;
  float v = 0.f;
  if (k < 100)               v = h[(long)s * 100 + k];
  else if (k < 100 + se)     v = es[e * (long)se + (k - 100)];
  else if (k < 200 + se)     v = h[(long)t * 100 + (k - 100 - se)];
  else if (k < 233 + se)     v = vn1[e * 33 + (k - 200 - se)];
  A[idx] = (__bf16)v;
}

__global__ void pack2_bf16(const float* __restrict__ X, int cx,
                           const float* __restrict__ Y, int cy,
                           __bf16* __restrict__ A, int Kp, long M) {
  long idx = blockIdx.x * (long)blockDim.x + threadIdx.x;
  if (idx >= M * Kp) return;
  long m = idx / Kp;
  int  k = (int)(idx % Kp);
  float v = 0.f;
  if (k < cx)            v = X[m * (long)cx + k];
  else if (k < cx + cy)  v = Y[m * (long)cy + (k - cx)];
  A[idx] = (__bf16)v;
}

__global__ void build_es_dec(const float* __restrict__ es, const int* __restrict__ src,
                             const int* __restrict__ dst, const int* __restrict__ seq,
                             const float* __restrict__ Ws, float* __restrict__ out, long E) {
  long idx = blockIdx.x * (long)blockDim.x + threadIdx.x;
  if (idx >= E * 52) return;
  long e = idx / 52;
  int  k = (int)(idx % 52);
  float v;
  if (k < 32) v = es[e * 32 + k];
  else {
    int s = src[e];
    v = (s < dst[e]) ? Ws[(long)seq[s] * 20 + (k - 32)] : 0.f;
  }
  out[idx] = v;
}

// ---------------------------------------------------------------------------
// Segment sum (atomics) + degree count
// ---------------------------------------------------------------------------
__global__ void count_dst_k(const int* __restrict__ dst, float* __restrict__ cnt, long E) {
  long i = blockIdx.x * (long)blockDim.x + threadIdx.x;
  if (i < E) atomicAdd(&cnt[dst[i]], 1.f);
}

__global__ void scatter_edges(const int* __restrict__ dst,
                              const float* __restrict__ ms, const float* __restrict__ mv,
                              float* __restrict__ dhs, float* __restrict__ dhv, long E) {
  long idx = blockIdx.x * (long)blockDim.x + threadIdx.x;
  if (idx >= E * 148) return;
  long e = idx / 148;
  int  c = (int)(idx % 148);
  int  d = dst[e];
  if (c < 100) atomicAdd(&dhs[(long)d * 100 + c], ms[e * 100 + c]);
  else         atomicAdd(&dhv[(long)d * 48 + (c - 100)], mv[e * 48 + (c - 100)]);
}

// ---------------------------------------------------------------------------
// Fused (optional residual/segment-mean) + GVP layernorm.  One wave per row.
// ---------------------------------------------------------------------------
__global__ __launch_bounds__(256)
void residual_ln(const float* __restrict__ s_in, const float* __restrict__ add_s,
                 const float* __restrict__ v_in, const float* __restrict__ add_v,
                 const float* __restrict__ cnt,
                 const float* __restrict__ w, const float* __restrict__ b,
                 float* __restrict__ s_out, float* __restrict__ v_out,
                 long Nn, int S, int V) {
  long node = (long)blockIdx.x * 8 + (threadIdx.x >> 5);
  int  lane = threadIdx.x & 31;
  if (node >= Nn) return;
  float inv = 1.f;
  if (cnt) inv = 1.f / fmaxf(cnt[node], 1.f);

  float sv[4];
  float sum = 0.f;
  #pragma unroll
  for (int i = 0; i < 4; ++i) {
    int c = lane + i * 32;
    float x = 0.f;
    if (c < S) {
      x = s_in[node * S + c];
      if (add_s) x += add_s[node * S + c] * inv;
    }
    sv[i] = x; sum += x;
  }
  for (int o = 16; o; o >>= 1) sum += __shfl_xor(sum, o, 32);
  float mu = sum / (float)S;
  float var = 0.f;
  #pragma unroll
  for (int i = 0; i < 4; ++i) {
    int c = lane + i * 32;
    if (c < S) { float d0 = sv[i] - mu; var += d0 * d0; }
  }
  for (int o = 16; o; o >>= 1) var += __shfl_xor(var, o, 32);
  float rstd = rsqrtf(var / (float)S + 1e-5f);
  #pragma unroll
  for (int i = 0; i < 4; ++i) {
    int c = lane + i * 32;
    if (c < S) s_out[node * S + c] = (sv[i] - mu) * rstd * w[c] + b[c];
  }

  float acc = 0.f;
  for (int j = lane; j < V; j += 32) {
    float n2 = 0.f;
    for (int d = 0; d < 3; ++d) {
      float x = v_in[(node * 3 + d) * (long)V + j];
      if (add_v) x += add_v[(node * 3 + d) * (long)V + j] * inv;
      n2 += x * x;
    }
    acc += fmaxf(n2, EPSF);
  }
  for (int o = 16; o; o >>= 1) acc += __shfl_xor(acc, o, 32);
  float rv = 1.f / sqrtf(acc / (float)V);
  for (int j = lane; j < V; j += 32)
    for (int d = 0; d < 3; ++d) {
      long off = (node * 3 + d) * (long)V + j;
      float x = v_in[off];
      if (add_v) x += add_v[off] * inv;
      v_out[off] = x * rv;
    }
}

// ---------------------------------------------------------------------------
// log_softmax over C classes, one wave per row
// ---------------------------------------------------------------------------
__global__ __launch_bounds__(256)
void log_softmax_k(const float* __restrict__ x, float* __restrict__ y, long Nn, int C) {
  long node = (long)blockIdx.x * 8 + (threadIdx.x >> 5);
  int  lane = threadIdx.x & 31;
  if (node >= Nn) return;
  float m = -1e30f;
  for (int c = lane; c < C; c += 32) m = fmaxf(m, x[node * C + c]);
  for (int o = 16; o; o >>= 1) m = fmaxf(m, __shfl_xor(m, o, 32));
  float s = 0.f;
  for (int c = lane; c < C; c += 32) s += expf(x[node * C + c] - m);
  for (int o = 16; o; o >>= 1) s += __shfl_xor(s, o, 32);
  float l = logf(s) + m;
  for (int c = lane; c < C; c += 32) y[node * C + c] = x[node * C + c] - l;
}

// ===========================================================================
// Host side
// ===========================================================================
struct LayerP {
  const float *ff0_wh, *ff0_b, *ff0_w, *ff0_wv;
  const float *ff1_wh, *ff1_b, *ff1_w, *ff1_wv;
  const float *ln0_b, *ln0_w, *ln1_b, *ln1_w;
  const float *m0_wh, *m0_b, *m0_w, *m0_wv;
  const float *m1_wh, *m1_b, *m1_w, *m1_wv;
  const float *m2_wh, *m2_b, *m2_w, *m2_wv;
};
struct LayerB { __bf16 *m0, *m1, *m2, *ff0, *ff1; };

static inline dim3 g1d(long n, int blk) { return dim3((unsigned)((n + blk - 1) / blk)); }
static inline dim3 g8(long n) { return dim3((unsigned)((n + 7) / 8)); }

extern "C" void kernel_launch(void* const* d_in, const int* in_sizes, int n_in,
                              void* d_out, int out_size, void* d_ws, size_t ws_size,
                              hipStream_t stream) {
  const float* node_s = (const float*)d_in[0];
  const float* node_v = (const float*)d_in[1];
  const float* edge_s = (const float*)d_in[2];
  const float* edge_v = (const float*)d_in[3];
  const int*   eidx   = (const int*)d_in[4];
  const int*   seq    = (const int*)d_in[5];
  const long Nn = in_sizes[0] / 6;
  const long E  = in_sizes[4] / 2;
  const int* src = eidx;
  const int* dst = eidx + E;

  // ---- param walk: jax tree order (dict keys sorted):
  // W_e, W_out, W_s, W_v, dec[3], enc[3], ln_e, ln_v; gvp dict: wh, ws_b, ws_w, wv
  int cur = 6;
  auto nf = [&]() { return (const float*)d_in[cur++]; };
  LayerP Ldec[3], Lenc[3];
  auto readLayer = [&](LayerP& L) {
    L.ff0_wh = nf(); L.ff0_b = nf(); L.ff0_w = nf(); L.ff0_wv = nf();
    L.ff1_wh = nf(); L.ff1_b = nf(); L.ff1_w = nf(); L.ff1_wv = nf();
    L.ln0_b = nf(); L.ln0_w = nf();
    L.ln1_b = nf(); L.ln1_w = nf();
    L.m0_wh = nf(); L.m0_b = nf(); L.m0_w = nf(); L.m0_wv = nf();
    L.m1_wh = nf(); L.m1_b = nf(); L.m1_w = nf(); L.m1_wv = nf();
    L.m2_wh = nf(); L.m2_b = nf(); L.m2_w = nf(); L.m2_wv = nf();
  };
  const float *We_wh = nf(), *We_b = nf(), *We_w = nf(), *We_wv = nf();
  const float *Wo_wh = nf(), *Wo_b = nf(), *Wo_w = nf();
  const float *Ws    = nf();
  const float *Wv_wh = nf(), *Wv_b = nf(), *Wv_w = nf(), *Wv_wv = nf();
  for (int i = 0; i < 3; ++i) readLayer(Ldec[i]);
  for (int i = 0; i < 3; ++i) readLayer(Lenc[i]);
  const float *lne_b = nf(), *lne_w = nf();
  const float *lnv_b = nf(), *lnv_w = nf();

  // ---- workspace bump allocator
  char* base = (char*)d_ws;
  size_t off = 0;
  auto alloc = [&](size_t bytes) -> void* {
    void* p = base + off;
    off = (off + bytes + 255) & ~(size_t)255;
    return p;
  };
  auto allocF = [&](long n) { return (float*)alloc((size_t)n * 4); };
  auto allocH = [&](long n) { return (__bf16*)alloc((size_t)n * 2); };

  // bf16 transposed/padded weight copies (GEMM B operands)
  auto cv = [&](const float* w, int K, int N) {
    int Kp = (K + 31) & ~31, Np = (N + 63) & ~63;
    long tot = (long)Np * Kp;
    __bf16* p = allocH(tot);
    prep_B_bf16<<<g1d(tot, 256), 256, 0, stream>>>(w, p, K, N, Kp, tot);
    return p;
  };
  __bf16* Wv_wB = cv(Wv_w, 22, 100);
  __bf16* We_wB = cv(We_w, 33, 32);
  __bf16* Wo_wB = cv(Wo_w, 116, 33);
  LayerB Bdec[3], Benc[3];
  for (int i = 0; i < 3; ++i) {
    Bdec[i].m0  = cv(Ldec[i].m0_w, 285, 100);
    Bdec[i].m1  = cv(Ldec[i].m1_w, 116, 100);
    Bdec[i].m2  = cv(Ldec[i].m2_w, 116, 100);
    Bdec[i].ff0 = cv(Ldec[i].ff0_w, 132, 400);
    Bdec[i].ff1 = cv(Ldec[i].ff1_w, 432, 100);
  }
  for (int i = 0; i < 3; ++i) {
    Benc[i].m0  = cv(Lenc[i].m0_w, 265, 100);
    Benc[i].m1  = cv(Lenc[i].m1_w, 116, 100);
    Benc[i].m2  = cv(Lenc[i].m2_w, 116, 100);
    Benc[i].ff0 = cv(Lenc[i].ff0_w, 132, 400);
    Benc[i].ff1 = cv(Lenc[i].ff1_w, 432, 100);
  }

  // node buffers
  float* cnt   = allocF(Nn);
  float* hs    = allocF(Nn * 100);
  float* hv    = allocF(Nn * 48);
  float* encs  = allocF(Nn * 100);
  float* encv  = allocF(Nn * 48);
  float* dhs   = allocF(Nn * 100);
  float* dhv   = allocF(Nn * 48);
  float* fs1   = allocF(Nn * 400);
  float* fsout = allocF(Nn * 100);
  float* vhf   = allocF(Nn * 96);
  float* vnf   = allocF(Nn * 32);
  float* fv1   = allocF(Nn * 96);
  float* vhf2  = allocF(Nn * 96);
  float* vnf2  = allocF(Nn * 32);
  float* fvout = allocF(Nn * 48);
  __bf16* An   = allocH(Nn * 448);
  // edge buffers
  float* es    = allocF(E * 32);
  float* ev    = allocF(E * 3);
  float* esd   = allocF(E * 52);
  float* mv    = allocF(E * 99);
  float* vh1   = allocF(E * 99);
  float* vn1   = allocF(E * 33);
  float* ms1   = allocF(E * 100);
  float* ms2   = allocF(E * 100);
  float* mvx   = allocF(E * 48);
  float* mvy   = allocF(E * 48);
  float* vh2   = allocF(E * 48);   // also vh3 / edge-embed vh
  float* vn2   = allocF(E * 16);   // also vn3 / edge-embed vn
  __bf16* A1   = allocH(E * 288);
  __bf16* A2   = allocH(E * 128);  // also edge-embed panel (Kp=64)

  auto gemm = [&](const __bf16* A, int lda, const __bf16* Bt, const float* bias,
                  float* C, int ldc, long M, int K, int N, int act) {
    int Kp = (K + 31) & ~31;
    int Np = (N + 63) & ~63;
    dim3 grid((unsigned)((M + 127) / 128), (unsigned)(Np / 64));
    size_t sh = (size_t)64 * Kp * sizeof(__bf16);
    gemm_bf16_wmma<<<grid, 256, sh, stream>>>(A, lda, Bt, bias, C, ldc, M, K, N, Kp, act);
  };

  // ---- degree counts (reused by every conv layer)
  hipMemsetAsync(cnt, 0, Nn * sizeof(float), stream);
  count_dst_k<<<g1d(E, 256), 256, 0, stream>>>(dst, cnt, E);

  // ---- node embedding: W_v then ln_v
  vchan_mm<<<g1d(Nn * 16, 256), 256, 0, stream>>>(node_v, 9, 1, 3, Wv_wh, vhf, vnf, Nn, 3, 16);
  pack2_bf16<<<g1d(Nn * 32, 256), 256, 0, stream>>>(node_s, 6, vnf, 16, An, 32, Nn);
  gemm(An, 32, Wv_wB, Wv_b, hs, 100, Nn, 22, 100, 0);
  vchan_out<<<g1d(Nn * 16, 256), 256, 0, stream>>>(vhf, Wv_wv, hv, Nn, 16, 16, 0);
  residual_ln<<<g8(Nn), 256, 0, stream>>>(hs, nullptr, hv, nullptr, nullptr,
                                          lnv_w, lnv_b, hs, hv, Nn, 100, 16);

  // ---- edge embedding: W_e then ln_e
  vchan_mm<<<g1d(E, 256), 256, 0, stream>>>(edge_v, 3, 1, 1, We_wh, vh2, vn2, E, 1, 1);
  pack2_bf16<<<g1d(E * 64, 256), 256, 0, stream>>>(edge_s, 32, vn2, 1, A2, 64, E);
  gemm(A2, 64, We_wB, We_b, es, 32, E, 33, 32, 0);
  vchan_out<<<g1d(E, 256), 256, 0, stream>>>(vh2, We_wv, ev, E, 1, 1, 0);
  residual_ln<<<g8(E), 256, 0, stream>>>(es, nullptr, ev, nullptr, nullptr,
                                         lne_w, lne_b, es, ev, E, 32, 1);

  // ---- one conv layer
  auto runLayer = [&](const LayerP& L, const LayerB& B, const float* esl, int se,
                      const float* eS, const float* eV) {
    const int K1 = 233 + se;
    build_msg_mv<<<g1d(E * 99, 256), 256, 0, stream>>>(src, dst, hv, eV, ev, mv, E);
    vchan_mm<<<g1d(E * 33, 256), 256, 0, stream>>>(mv, 99, 33, 1, L.m0_wh, vh1, vn1, E, 33, 33);
    build_msg_A<<<g1d(E * 288, 256), 256, 0, stream>>>(src, dst, hs, eS, esl, se, vn1, A1, 288, E);
    gemm(A1, 288, B.m0, L.m0_b, ms1, 100, E, K1, 100, 1);
    vchan_out<<<g1d(E * 16, 256), 256, 0, stream>>>(vh1, L.m0_wv, mvx, E, 33, 16, 1);
    vchan_mm<<<g1d(E * 16, 256), 256, 0, stream>>>(mvx, 48, 16, 1, L.m1_wh, vh2, vn2, E, 16, 16);
    pack2_bf16<<<g1d(E * 128, 256), 256, 0, stream>>>(ms1, 100, vn2, 16, A2, 128, E);
    gemm(A2, 128, B.m1, L.m1_b, ms2, 100, E, 116, 100, 1);
    vchan_out<<<g1d(E * 16, 256), 256, 0, stream>>>(vh2, L.m1_wv, mvy, E, 16, 16, 1);
    vchan_mm<<<g1d(E * 16, 256), 256, 0, stream>>>(mvy, 48, 16, 1, L.m2_wh, vh2, vn2, E, 16, 16);
    pack2_bf16<<<g1d(E * 128, 256), 256, 0, stream>>>(ms2, 100, vn2, 16, A2, 128, E);
    gemm(A2, 128, B.m2, L.m2_b, ms1, 100, E, 116, 100, 0);           // ms3 -> ms1
    vchan_out<<<g1d(E * 16, 256), 256, 0, stream>>>(vh2, L.m2_wv, mvx, E, 16, 16, 0);
    hipMemsetAsync(dhs, 0, (size_t)Nn * 100 * 4, stream);
    hipMemsetAsync(dhv, 0, (size_t)Nn * 48 * 4, stream);
    scatter_edges<<<g1d(E * 148, 256), 256, 0, stream>>>(dst, ms1, mvx, dhs, dhv, E);
    residual_ln<<<g8(Nn), 256, 0, stream>>>(hs, dhs, hv, dhv, cnt,
                                            L.ln0_w, L.ln0_b, hs, hv, Nn, 100, 16);
    // feed-forward
    vchan_mm<<<g1d(Nn * 32, 256), 256, 0, stream>>>(hv, 48, 16, 1, L.ff0_wh, vhf, vnf, Nn, 16, 32);
    pack2_bf16<<<g1d(Nn * 160, 256), 256, 0, stream>>>(hs, 100, vnf, 32, An, 160, Nn);
    gemm(An, 160, B.ff0, L.ff0_b, fs1, 400, Nn, 132, 400, 1);
    vchan_out<<<g1d(Nn * 32, 256), 256, 0, stream>>>(vhf, L.ff0_wv, fv1, Nn, 32, 32, 1);
    vchan_mm<<<g1d(Nn * 32, 256), 256, 0, stream>>>(fv1, 96, 32, 1, L.ff1_wh, vhf2, vnf2, Nn, 32, 32);
    pack2_bf16<<<g1d(Nn * 448, 256), 256, 0, stream>>>(fs1, 400, vnf2, 32, An, 448, Nn);
    gemm(An, 448, B.ff1, L.ff1_b, fsout, 100, Nn, 432, 100, 0);
    vchan_out<<<g1d(Nn * 16, 256), 256, 0, stream>>>(vhf2, L.ff1_wv, fvout, Nn, 32, 16, 0);
    residual_ln<<<g8(Nn), 256, 0, stream>>>(hs, fsout, hv, fvout, nullptr,
                                            L.ln1_w, L.ln1_b, hs, hv, Nn, 100, 16);
  };

  // ---- encoder
  for (int l = 0; l < 3; ++l) runLayer(Lenc[l], Benc[l], es, 32, nullptr, nullptr);
  hipMemcpyAsync(encs, hs, (size_t)Nn * 100 * 4, hipMemcpyDeviceToDevice, stream);
  hipMemcpyAsync(encv, hv, (size_t)Nn * 48 * 4, hipMemcpyDeviceToDevice, stream);

  // ---- decoder edge features + decoder layers
  build_es_dec<<<g1d(E * 52, 256), 256, 0, stream>>>(es, src, dst, seq, Ws, esd, E);
  for (int l = 0; l < 3; ++l) runLayer(Ldec[l], Bdec[l], esd, 52, encs, encv);

  // ---- output head: W_out GVP (vo=0) -> logits -> log_softmax
  vchan_mm<<<g1d(Nn * 16, 256), 256, 0, stream>>>(hv, 48, 16, 1, Wo_wh, vhf, vnf, Nn, 16, 16);
  pack2_bf16<<<g1d(Nn * 128, 256), 256, 0, stream>>>(hs, 100, vnf, 16, An, 128, Nn);
  float* logits = (float*)d_out + Nn * 33;   // second tuple element
  gemm(An, 128, Wo_wB, Wo_b, logits, 33, Nn, 116, 33, 0);
  log_softmax_k<<<g8(Nn), 256, 0, stream>>>(logits, (float*)d_out, Nn, 33);
}
